// GCN_88519275970865
// MI455X (gfx1250) — matrix operand order repaired and checked by reading, not compiled, not run on hardware
//
#include <hip/hip_runtime.h>
#include <hip/hip_bf16.h>

// ---------------------------------------------------------------------------
// GCN forward: 2 layers of
//   agg[r] = adj[r] @ emb          (R,N,D)   <-- dominant: 34.4 GFLOP/layer
//   tmp[r] = agg[r] @ rel[r]^T     (R,N,D)
//   emb    = relu(mean_r tmp)      (N,D)
// R=8, N=4096, D=128.  adj streamed f32+NT (512MB/layer ~= 22us at 23.3TB/s),
// math in v_wmma_f32_16x16x32_f16.  Register-prefetch + double-buffered LDS
// pipeline, enforced with sched_barrier so global loads stay hoisted above
// the WMMA block (one loadcnt wait per iteration, fully overlapped).
// ---------------------------------------------------------------------------

#define RREL 8
#define NENT 4096
#define DEMB 128
#define BK   32
#define LSTR 40   // LDS row stride in halves (32 + 8 pad) = 80B, 16B aligned

typedef _Float16 v16h __attribute__((ext_vector_type(16)));
typedef _Float16 v8h  __attribute__((ext_vector_type(8)));
typedef _Float16 v4h  __attribute__((ext_vector_type(4)));
typedef float    v8f  __attribute__((ext_vector_type(8)));
typedef float    v4f  __attribute__((ext_vector_type(4)));

// sched_group_barrier masks
#define SGB_WMMA 0x008
#define SGB_DSRD 0x100

// A fragment (16x32 f16): lane<16 -> row=lane,   K = 0..15
//                         lane>=16-> row=lane-16,K = 16..31
__device__ __forceinline__ v16h load_frag_a(const _Float16* lds, int rowBase, int lane) {
  const int half = lane >> 4;
  const _Float16* p = lds + (rowBase + (lane & 15)) * LSTR + half * 16;
  const v8h lo = *(const v8h*)(p);
  const v8h hi = *(const v8h*)(p + 8);
  return __builtin_shufflevector(lo, hi, 0,1,2,3,4,5,6,7,8,9,10,11,12,13,14,15);
}

// B fragment (32x16 f16): lane<16 -> col=lane,   K = 0..15
//                         lane>=16-> col=lane-16,K = 16..31
// LDS B tile stored col-major: lB[col][k]
__device__ __forceinline__ v16h load_frag_b(const _Float16* lds, int colBase, int lane) {
  const int half = lane >> 4;
  const _Float16* p = lds + (colBase + (lane & 15)) * LSTR + half * 16;
  const v8h lo = *(const v8h*)(p);
  const v8h hi = *(const v8h*)(p + 8);
  return __builtin_shufflevector(lo, hi, 0,1,2,3,4,5,6,7,8,9,10,11,12,13,14,15);
}

// ---------------------------------------------------------------------------
// GEMM1: aggH[r] = f16( adj[r] (NxN f32) @ embT^T )   -> (N x D) f16
//   grid = (N/128, R), block = 256 (8 waves).  Tile 128(M) x 128(N=D).
// ---------------------------------------------------------------------------
__global__ __launch_bounds__(256) void gemm1_kernel(
    const float* __restrict__ adj, const _Float16* __restrict__ embT,
    _Float16* __restrict__ aggH) {
  __shared__ __align__(16) _Float16 lA[2][128 * LSTR];
  __shared__ __align__(16) _Float16 lB[2][128 * LSTR];

  const int tid   = threadIdx.x;
  const int lane  = tid & 31;
  const int wave  = tid >> 5;
  const int waveM = wave >> 2;          // 0..1  (rows)
  const int waveN = wave & 3;           // 0..3  (cols)
  const int nBase = blockIdx.x * 128;   // output row tile
  const int r     = blockIdx.y;

  v8f acc[4][2];
  for (int i = 0; i < 4; ++i)
    for (int j = 0; j < 2; ++j)
      for (int e = 0; e < 8; ++e) acc[i][j][e] = 0.0f;

  const size_t adjR = (size_t)r * NENT * NENT;
  const int arow = tid >> 3;            // 0..31 : A staging base row
  const int acol = (tid & 7) * 4;       // f32 column
  const int brow = tid >> 1;            // 0..127: B staging d row
  const int boff = (tid & 1) * 16;      // halves

  v4f ga[4];                            // prefetched adj f32
  v8h gb[2];                            // prefetched embT f16

  auto load_tile = [&](int kBase) {
    #pragma unroll
    for (int i = 0; i < 4; ++i) {
      const int row = arow + i * 32;
      ga[i] = __builtin_nontemporal_load(
          (const v4f*)(adj + adjR + (size_t)(nBase + row) * NENT + kBase + acol));
    }
    const _Float16* p = embT + (size_t)brow * NENT + kBase + boff;
    gb[0] = *(const v8h*)(p);
    gb[1] = *(const v8h*)(p + 8);
  };
  auto store_tile = [&](int buf) {
    #pragma unroll
    for (int i = 0; i < 4; ++i) {
      const int row = arow + i * 32;
      v4h h;
      h[0] = (_Float16)ga[i][0]; h[1] = (_Float16)ga[i][1];
      h[2] = (_Float16)ga[i][2]; h[3] = (_Float16)ga[i][3];
      *(v4h*)&lA[buf][row * LSTR + acol] = h;
    }
    *(v8h*)&lB[buf][brow * LSTR + boff]     = gb[0];
    *(v8h*)&lB[buf][brow * LSTR + boff + 8] = gb[1];
  };

  const int steps = NENT / BK;          // 128
  load_tile(0);
  store_tile(0);
  __syncthreads();

  for (int i = 0; i < steps; ++i) {
    const int cur = i & 1;
    const int inext = (i + 1 < steps) ? (i + 1) : i;   // branchless: last iter reloads
    load_tile(inext * BK);                             // issue early, drain under WMMA
    __builtin_amdgcn_sched_barrier(0);

    v16h af[4], bf[2];
    #pragma unroll
    for (int mt = 0; mt < 4; ++mt)
      af[mt] = load_frag_a(lA[cur], waveM * 64 + mt * 16, lane);
    #pragma unroll
    for (int nt = 0; nt < 2; ++nt)
      bf[nt] = load_frag_b(lB[cur], waveN * 32 + nt * 16, lane);
    #pragma unroll
    for (int mt = 0; mt < 4; ++mt)
      #pragma unroll
      for (int nt = 0; nt < 2; ++nt)
        acc[mt][nt] = __builtin_amdgcn_wmma_f32_16x16x32_f16(
            false, af[mt], false, bf[nt], (short)0, acc[mt][nt], false, false);
    // request grouping: all 12 frag DS reads, then the 8 WMMAs
    __builtin_amdgcn_sched_group_barrier(SGB_DSRD, 12, 0);
    __builtin_amdgcn_sched_group_barrier(SGB_WMMA, 8, 0);
    __builtin_amdgcn_sched_barrier(0);

    store_tile(1 - cur);
    __syncthreads();
  }

  // ---- epilogue: store f16 agg tile (row-major [r][n][d]) ----
  const int hi8 = (lane >> 4) * 8;
  const int col = lane & 15;
  #pragma unroll
  for (int mt = 0; mt < 4; ++mt)
    #pragma unroll
    for (int nt = 0; nt < 2; ++nt) {
      const int rowT = nBase + waveM * 64 + mt * 16 + hi8;
      const int colT = waveN * 32 + nt * 16 + col;
      #pragma unroll
      for (int g = 0; g < 8; ++g)
        aggH[((size_t)r * NENT + rowT + g) * DEMB + colT] =
            (_Float16)acc[mt][nt][g];
    }
}

// ---------------------------------------------------------------------------
// GEMM2: out[n,d] = relu( (1/R) * sum_r sum_e aggH[r][n][e] * relH[r][d][e] )
//   K = R*D = 1024 fused accumulation.  grid = N/128, block = 256.
// ---------------------------------------------------------------------------
__global__ __launch_bounds__(256) void gemm2_kernel(
    const _Float16* __restrict__ aggH, const _Float16* __restrict__ relH,
    float* __restrict__ out) {
  __shared__ __align__(16) _Float16 lA[2][128 * LSTR];
  __shared__ __align__(16) _Float16 lB[2][128 * LSTR];

  const int tid   = threadIdx.x;
  const int lane  = tid & 31;
  const int wave  = tid >> 5;
  const int waveM = wave >> 2;
  const int waveN = wave & 3;
  const int nBase = blockIdx.x * 128;

  v8f acc[4][2];
  for (int i = 0; i < 4; ++i)
    for (int j = 0; j < 2; ++j)
      for (int e = 0; e < 8; ++e) acc[i][j][e] = 0.0f;

  const int srow = tid >> 1;          // 0..127
  const int soff = (tid & 1) * 16;    // halves

  v8h gaH[2], gbH[2];

  auto load_tile = [&](int kk) {
    const int rr = kk >> 7;           // kk / 128
    const int e  = kk & 127;
    const _Float16* pa = aggH + ((size_t)rr * NENT + nBase + srow) * DEMB + e + soff;
    gaH[0] = *(const v8h*)(pa);
    gaH[1] = *(const v8h*)(pa + 8);
    const _Float16* pb = relH + ((size_t)rr * DEMB + srow) * DEMB + e + soff;
    gbH[0] = *(const v8h*)(pb);
    gbH[1] = *(const v8h*)(pb + 8);
  };
  auto store_tile = [&](int buf) {
    *(v8h*)&lA[buf][srow * LSTR + soff]     = gaH[0];
    *(v8h*)&lA[buf][srow * LSTR + soff + 8] = gaH[1];
    *(v8h*)&lB[buf][srow * LSTR + soff]     = gbH[0];
    *(v8h*)&lB[buf][srow * LSTR + soff + 8] = gbH[1];
  };

  const int steps = (RREL * DEMB) / BK;   // 32
  load_tile(0);
  store_tile(0);
  __syncthreads();

  for (int i = 0; i < steps; ++i) {
    const int cur = i & 1;
    const int inext = (i + 1 < steps) ? (i + 1) : i;
    load_tile(inext * BK);
    __builtin_amdgcn_sched_barrier(0);

    v16h af[4], bf[2];
    #pragma unroll
    for (int mt = 0; mt < 4; ++mt)
      af[mt] = load_frag_a(lA[cur], waveM * 64 + mt * 16, lane);
    #pragma unroll
    for (int nt = 0; nt < 2; ++nt)
      bf[nt] = load_frag_b(lB[cur], waveN * 32 + nt * 16, lane);
    #pragma unroll
    for (int mt = 0; mt < 4; ++mt)
      #pragma unroll
      for (int nt = 0; nt < 2; ++nt)
        acc[mt][nt] = __builtin_amdgcn_wmma_f32_16x16x32_f16(
            false, af[mt], false, bf[nt], (short)0, acc[mt][nt], false, false);
    __builtin_amdgcn_sched_group_barrier(SGB_DSRD, 12, 0);
    __builtin_amdgcn_sched_group_barrier(SGB_WMMA, 8, 0);
    __builtin_amdgcn_sched_barrier(0);

    store_tile(1 - cur);
    __syncthreads();
  }

  // ---- epilogue: mean over r (x 1/8), relu, store f32 ----
  const int hi8 = (lane >> 4) * 8;
  const int col = lane & 15;
  #pragma unroll
  for (int mt = 0; mt < 4; ++mt)
    #pragma unroll
    for (int nt = 0; nt < 2; ++nt) {
      const int rowT = nBase + waveM * 64 + mt * 16 + hi8;
      const int colT = waveN * 32 + nt * 16 + col;
      #pragma unroll
      for (int g = 0; g < 8; ++g) {
        const float v = acc[mt][nt][g] * 0.125f;
        out[(size_t)(rowT + g) * DEMB + colT] = v > 0.0f ? v : 0.0f;
      }
    }
}

// ---------------------------------------------------------------------------
// Small helpers: f32 -> f16 convert, and f32 [N][D] -> f16 [D][N] transpose.
// ---------------------------------------------------------------------------
__global__ __launch_bounds__(256) void cvt_f16_kernel(
    const float* __restrict__ src, _Float16* __restrict__ dst, int count) {
  const int i = blockIdx.x * 256 + threadIdx.x;
  if (i < count) dst[i] = (_Float16)src[i];
}

__global__ __launch_bounds__(256) void transpose_f16_kernel(
    const float* __restrict__ src, _Float16* __restrict__ dst) {
  const int i = blockIdx.x * 256 + threadIdx.x;   // over N*D
  const int n = i >> 7;       // / D
  const int d = i & 127;      // % D
  dst[(size_t)d * NENT + n] = (_Float16)src[i];
}

// ---------------------------------------------------------------------------
extern "C" void kernel_launch(void* const* d_in, const int* in_sizes, int n_in,
                              void* d_out, int out_size, void* d_ws, size_t ws_size,
                              hipStream_t stream) {
  (void)in_sizes; (void)n_in; (void)out_size; (void)ws_size;

  const float* adj = (const float*)d_in[0];   // (R,N,N) f32
  const float* ent = (const float*)d_in[1];   // (N,D)   f32
  const float* rel = (const float*)d_in[2];   // (R,D,D) f32
  float*       out = (float*)d_out;           // (N,D)   f32

  char* ws = (char*)d_ws;
  _Float16* embT    = (_Float16*)(ws);                                  // 1 MB : f16 [D][N]
  _Float16* aggH    = (_Float16*)(ws + (1u << 20));                     // 8 MB : f16 [R][N][D]
  _Float16* relH    = (_Float16*)(ws + (1u << 20) + (8u << 20));        // 256KB: f16 [R][D][D]
  float*    embNext = (float*)   (ws + (1u << 20) + (8u << 20) + (256u << 10)); // 2 MB

  // one-time converts
  cvt_f16_kernel<<<(RREL * DEMB * DEMB) / 256, 256, 0, stream>>>(rel, relH, RREL * DEMB * DEMB);
  transpose_f16_kernel<<<(NENT * DEMB) / 256, 256, 0, stream>>>(ent, embT);

  // layer 0
  gemm1_kernel<<<dim3(NENT / 128, RREL), 256, 0, stream>>>(adj, embT, aggH);
  gemm2_kernel<<<NENT / 128, 256, 0, stream>>>(aggH, relH, embNext);
  transpose_f16_kernel<<<(NENT * DEMB) / 256, 256, 0, stream>>>(embNext, embT);

  // layer 1 (final -> d_out)
  gemm1_kernel<<<dim3(NENT / 128, RREL), 256, 0, stream>>>(adj, embT, aggH);
  gemm2_kernel<<<NENT / 128, 256, 0, stream>>>(aggH, relH, out);
}
